// FAVOR_4406636445947
// MI455X (gfx1250) — matrix operand order repaired and checked by reading, not compiled, not run on hardware
//
#include <hip/hip_runtime.h>
#include <hip/hip_bf16.h>

typedef __attribute__((ext_vector_type(16))) __bf16 v16bf;
typedef __attribute__((ext_vector_type(8)))  __bf16 v8bf;
typedef __attribute__((ext_vector_type(8)))  float  v8f;

#define B_  4
#define S_  2048
#define CK  256
#define M_  128
#define DV  64
#define CH  64   // sequence chunk

// ---------------------------------------------------------------------------
// Fragment helpers (CDNA5 wave32 WMMA 16x16x32 bf16; this file's contiguous k order, same for A and B)
// A (16x32, 16-bit): lane m = lane&15, half h = lane>>4;
//   packed elems 0..15 -> K = k0 + 16h + 0..15
// ---------------------------------------------------------------------------
__device__ __forceinline__ v16bf fragA_row(const __bf16* p, int ld, int row0,
                                           int k0, int lane) {
  int r = row0 + (lane & 15);
  int h = (lane >> 4) & 1;
  const __bf16* q = p + r * ld + k0 + 16 * h;
  v8bf lo = *(const v8bf*)(q);
  v8bf hi = *(const v8bf*)(q + 8);
  v16bf out;
#pragma unroll
  for (int i = 0; i < 8; ++i) { out[i] = lo[i]; out[8 + i] = hi[i]; }
  return out;
}

// B (32x16, 16-bit): lane n = lane&15, half h = lane>>4; 16 contiguous K per
// lane starting at k0 + 16h.  Source stored "per-column contiguous in K":
// element (k, n) at p[n*ld + k]  -> two 16-byte LDS loads.
__device__ __forceinline__ v16bf fragB_col(const __bf16* p, int ld, int n0,
                                           int k0, int lane) {
  int n = n0 + (lane & 15);
  int h = (lane >> 4) & 1;
  const __bf16* q = p + n * ld + k0 + 16 * h;
  v8bf lo = *(const v8bf*)(q);
  v8bf hi = *(const v8bf*)(q + 8);
  v16bf out;
#pragma unroll
  for (int i = 0; i < 8; ++i) { out[i] = lo[i]; out[8 + i] = hi[i]; }
  return out;
}

// B fragment gathered from a row-major [k][n] LDS array.
__device__ __forceinline__ v16bf fragB_gather(const __bf16* p, int ld, int n0,
                                              int k0, int lane) {
  int n = n0 + (lane & 15);
  int h = (lane >> 4) & 1;
  v16bf out;
#pragma unroll
  for (int i = 0; i < 16; ++i) out[i] = p[(k0 + 16 * h + i) * ld + n];
  return out;
}

__device__ __forceinline__ v8f wmma_bf16(v16bf a, v16bf b, v8f c) {
  return __builtin_amdgcn_wmma_f32_16x16x32_bf16(false, a, false, b,
                                                 (short)0, c, false, false);
}

// ---------------------------------------------------------------------------
// Kernel 1: phi = relu(features @ X) + 1e-3 for X in {keys, queries}.
// grid = 2(src) x 4(b) x 32(s-tiles of 64); block = 128 (4 waves).
// Output bf16 layout: phiws[src][b][m][s].
// ---------------------------------------------------------------------------
__global__ __launch_bounds__(128) void phi_proj_kernel(
    const float* __restrict__ keys, const float* __restrict__ queries,
    const float* __restrict__ features, __bf16* __restrict__ phiws) {
  __shared__ __align__(16) __bf16 lF[M_ * 64];  // features panel [m][k]
  __shared__ __align__(16) __bf16 lX[64 * 64];  // X panel, stored [s][k]

  int bx = blockIdx.x;
  int st = bx & 31;
  int b = (bx >> 5) & 3;
  int src = bx >> 7;  // 0 = keys, 1 = queries
  int s0 = st * CH;
  const float* X = (src ? queries : keys) + (size_t)b * CK * S_;

  int tid = threadIdx.x;
  int lane = tid & 31, wv = tid >> 5;
  const v8f vzero = {0.f, 0.f, 0.f, 0.f, 0.f, 0.f, 0.f, 0.f};

  v8f acc[2][4];
#pragma unroll
  for (int i = 0; i < 2; ++i)
#pragma unroll
    for (int j = 0; j < 4; ++j) acc[i][j] = vzero;

  for (int k0 = 0; k0 < CK; k0 += 64) {
    {  // features panel: one row per thread, contiguous
      int r = tid;
      const float* g = features + (size_t)r * CK + k0;
      for (int j = 0; j < 64; ++j) lF[r * 64 + j] = (__bf16)g[j];
    }
    {  // X panel, transposed into [s][k]
      int kk = tid >> 1;
      int jh = (tid & 1) * 32;
      const float* g = X + (size_t)(k0 + kk) * S_ + s0 + jh;
      for (int j = 0; j < 32; ++j) lX[(jh + j) * 64 + kk] = (__bf16)g[j];
    }
    __syncthreads();
#pragma unroll
    for (int ks = 0; ks < 2; ++ks) {
      v16bf a0 = fragA_row(lF, 64, 32 * wv, 32 * ks, lane);
      v16bf a1 = fragA_row(lF, 64, 32 * wv + 16, 32 * ks, lane);
#pragma unroll
      for (int nt = 0; nt < 4; ++nt) {
        v16bf bb = fragB_col(lX, 64, 16 * nt, 32 * ks, lane);
        acc[0][nt] = wmma_bf16(a0, bb, acc[0][nt]);
        acc[1][nt] = wmma_bf16(a1, bb, acc[1][nt]);
      }
    }
    __syncthreads();
  }

  // epilogue: relu + kernel_eps, store bf16 to workspace
  int h = (lane >> 4) & 1, col = lane & 15;
  __bf16* outp = phiws + (size_t)(src * B_ + b) * M_ * S_;
#pragma unroll
  for (int mtl = 0; mtl < 2; ++mtl)
#pragma unroll
    for (int nt = 0; nt < 4; ++nt) {
      int s = s0 + 16 * nt + col;
#pragma unroll
      for (int v = 0; v < 8; ++v) {
        int m = 32 * wv + 16 * mtl + v + 8 * h;
        float pv = fmaxf(acc[mtl][nt][v], 0.f) + 1e-3f;
        outp[(size_t)m * S_ + s] = (__bf16)pv;
      }
    }
}

// ---------------------------------------------------------------------------
// Kernel 2: chunked causal linear-attention scan. One block per batch,
// 4 waves; wave w owns the 16-wide d-slice [16w,16w+16) of the running state
// S (128 x 64) in 8 f32 accumulator fragments.
// ---------------------------------------------------------------------------
__global__ __launch_bounds__(128) void favor_scan_kernel(
    const float* __restrict__ values, const __bf16* __restrict__ phiws,
    float* __restrict__ out) {
  __shared__ __align__(16) __bf16 lQ[CH * M_];  // phiQ [s][m]; reused as P [s][t]
  __shared__ __align__(16) __bf16 lK[M_ * CH];  // phiK [m][t]
  __shared__ __align__(16) __bf16 lV[DV * CH];  // V    [d][t]
  __shared__ __align__(16) __bf16 lS[DV * M_];  // state bf16 [d][m]
  __shared__ float zf[M_];
  __shared__ float qz[CH];
  __shared__ float rs[CH];

  int b = blockIdx.x;
  int tid = threadIdx.x, lane = tid & 31, wv = tid >> 5;
  int h = (lane >> 4) & 1, col = lane & 15;

  const __bf16* phiK = phiws + (size_t)(0 * B_ + b) * M_ * S_;
  const __bf16* phiQ = phiws + (size_t)(1 * B_ + b) * M_ * S_;
  const float* Vb = values + (size_t)b * DV * S_;
  float* outb = out + (size_t)b * DV * S_;

  const v8f vzero = {0.f, 0.f, 0.f, 0.f, 0.f, 0.f, 0.f, 0.f};
  v8f Sacc[8];
#pragma unroll
  for (int i = 0; i < 8; ++i) Sacc[i] = vzero;

  zf[tid] = 0.f;
  for (int i = tid; i < DV * M_; i += 128) lS[i] = (__bf16)0.f;
  __syncthreads();

  for (int s0 = 0; s0 < S_; s0 += CH) {
    // ---- load phase ----
    {
      int m = tid;
      const __bf16* gq = phiQ + (size_t)m * S_ + s0;
      for (int j = 0; j < CH; ++j) lQ[j * M_ + m] = gq[j];  // transpose -> [s][m]
      const v8bf* gk8 = (const v8bf*)(phiK + (size_t)m * S_ + s0);
      v8bf* lk8 = (v8bf*)(lK + m * CH);
#pragma unroll
      for (int j = 0; j < 8; ++j) lk8[j] = gk8[j];          // straight [m][t]
    }
    {
      int d = tid >> 1;
      int jh = (tid & 1) * 32;
      const float* g = Vb + (size_t)d * S_ + s0 + jh;
      for (int j = 0; j < 32; ++j) lV[d * CH + jh + j] = (__bf16)g[j];
    }
    __syncthreads();

    // qz[s] = phi_q[s] . z_prev
    if (tid < CH) {
      float a = 0.f;
      const __bf16* row = lQ + tid * M_;
      for (int m = 0; m < M_; ++m) a += (float)row[m] * zf[m];
      qz[tid] = a;
    }

    // ---- step 1: out_inter = phiQ @ S_prev (wave's d-slice) ----
    v8f Oacc[4];
#pragma unroll
    for (int i = 0; i < 4; ++i) Oacc[i] = vzero;
#pragma unroll
    for (int ks = 0; ks < 4; ++ks) {
      v16bf bS = fragB_col(lS, M_, 16 * wv, 32 * ks, lane);
#pragma unroll
      for (int rt = 0; rt < 4; ++rt) {
        v16bf a = fragA_row(lQ, M_, 16 * rt, 32 * ks, lane);
        Oacc[rt] = wmma_bf16(a, bS, Oacc[rt]);
      }
    }

    // ---- step 2: P strip = phiQ[16w:16w+16] @ phiK^T (causal: tiles j<=w) ----
    v8f Pacc[4];
#pragma unroll
    for (int j = 0; j < 4; ++j) Pacc[j] = vzero;
#pragma unroll
    for (int j = 0; j < 4; ++j) {
      if (j <= wv) {
#pragma unroll
        for (int ks = 0; ks < 4; ++ks) {
          v16bf a = fragA_row(lQ, M_, 16 * wv, 32 * ks, lane);
          v16bf bb = fragB_gather(lK, CH, 16 * j, 32 * ks, lane);
          Pacc[j] = wmma_bf16(a, bb, Pacc[j]);
        }
      }
    }
    __syncthreads();  // phiQ / lS reads done; qz written

    // ---- step 3: causal-mask P, store bf16 into lQ region; update z ----
    {
      __bf16* lP = lQ;
#pragma unroll
      for (int j = 0; j < 4; ++j) {
        int t = 16 * j + col;
#pragma unroll
        for (int v = 0; v < 8; ++v) {
          int s = 16 * wv + v + 8 * h;
          float pv = (t <= s) ? Pacc[j][v] : 0.f;
          lP[s * CH + t] = (__bf16)pv;
        }
      }
    }
    {  // z[m] += sum_t phiK[m][t]
      float a = zf[tid];
      const __bf16* row = lK + tid * CH;
      for (int t = 0; t < CH; ++t) a += (float)row[t];
      zf[tid] = a;
    }
    __syncthreads();  // masked P visible; z updated

    // rs[s] = row sum of masked P (intra normalizer)
    if (tid < CH) {
      float a = 0.f;
      const __bf16* row = lQ + tid * CH;
      for (int t = 0; t < CH; ++t) a += (float)row[t];
      rs[tid] = a;
    }
    __syncthreads();  // rs visible

    // ---- steps 4+5: out += P @ V-slice ; S += phiK(as [m][t]) @ V-slice ----
    v16bf bV0 = fragB_col(lV, CH, 16 * wv, 0, lane);
    v16bf bV1 = fragB_col(lV, CH, 16 * wv, 32, lane);
#pragma unroll
    for (int rt = 0; rt < 4; ++rt) {
      v16bf a0 = fragA_row(lQ, CH, 16 * rt, 0, lane);
      v16bf a1 = fragA_row(lQ, CH, 16 * rt, 32, lane);
      Oacc[rt] = wmma_bf16(a0, bV0, Oacc[rt]);
      Oacc[rt] = wmma_bf16(a1, bV1, Oacc[rt]);
    }
#pragma unroll
    for (int mt = 0; mt < 8; ++mt) {
      v16bf a0 = fragA_row(lK, CH, 16 * mt, 0, lane);
      v16bf a1 = fragA_row(lK, CH, 16 * mt, 32, lane);
      Sacc[mt] = wmma_bf16(a0, bV0, Sacc[mt]);
      Sacc[mt] = wmma_bf16(a1, bV1, Sacc[mt]);
    }

    // ---- normalize + write output (layout (b, d, s)) ----
    {
      int d = 16 * wv + col;
#pragma unroll
      for (int rt = 0; rt < 4; ++rt) {
#pragma unroll
        for (int v = 0; v < 8; ++v) {
          int s = 16 * rt + v + 8 * h;
          float norm = qz[s] + rs[s];
          outb[(size_t)d * S_ + s0 + s] = Oacc[rt][v] / norm;
        }
      }
    }

    // ---- mirror updated state to LDS bf16 for next chunk's step 1 ----
    {
      int d = 16 * wv + col;
#pragma unroll
      for (int mt = 0; mt < 8; ++mt) {
#pragma unroll
        for (int v = 0; v < 8; ++v) {
          int m = 16 * mt + v + 8 * h;
          lS[d * M_ + m] = (__bf16)Sacc[mt][v];
        }
      }
    }
    __syncthreads();  // lS ready; all reads of lQ/lK/lV complete
  }
}

// ---------------------------------------------------------------------------
extern "C" void kernel_launch(void* const* d_in, const int* in_sizes, int n_in,
                              void* d_out, int out_size, void* d_ws,
                              size_t ws_size, hipStream_t stream) {
  const float* keys = (const float*)d_in[0];
  const float* values = (const float*)d_in[1];
  const float* queries = (const float*)d_in[2];
  const float* features = (const float*)d_in[3];
  __bf16* phiws = (__bf16*)d_ws;  // 2 * 4 * 128 * 2048 bf16 = 4 MB

  phi_proj_kernel<<<dim3(256), dim3(128), 0, stream>>>(keys, queries, features,
                                                       phiws);
  favor_scan_kernel<<<dim3(4), dim3(128), 0, stream>>>(values, phiws,
                                                       (float*)d_out);
}
